// SpatialAttentionLayer_12876311954177
// MI455X (gfx1250) — matrix operand order repaired and checked by reading, not compiled, not run on hardware
//
#include <hip/hip_runtime.h>

typedef __attribute__((ext_vector_type(16))) _Float16 v16h;
typedef __attribute__((ext_vector_type(8)))  float    v8f;

#define IN_F   128
#define OUT_F  128
#define LOG2E  1.4426950408889634f
#define JSPLIT 4   // j-dimension split: 4x wave parallelism, partials add up

__device__ __forceinline__ float lrelu(float x) {
    // slope 0.2 < 1  =>  leaky_relu(x) == max(x, 0.2*x)
    return fmaxf(x, 0.2f * x);
}

// ---------------------------------------------------------------------------
// Kernel 0: zero the accumulation buffers (out and Z) — run every call so the
// atomic-add accumulation is deterministic across graph replays.
// ---------------------------------------------------------------------------
extern "C" __global__ void zero_bufs(float* __restrict__ out,
                                     float* __restrict__ Z,
                                     int nout, int nz) {
    const int t = blockIdx.x * 256 + threadIdx.x;
    if (t < nout) out[t] = 0.f;
    if (t < nz)   Z[t]   = 0.f;
}

// ---------------------------------------------------------------------------
// Kernel 1: Wh = h @ W   (N x 128) = (N x 128)(128 x 128).  0.27 GFLOP, trivial.
// ---------------------------------------------------------------------------
extern "C" __global__ void wh_gemm(const float* __restrict__ h,
                                   const float* __restrict__ W,
                                   float* __restrict__ Wh) {
    __shared__ float hrow[IN_F];
    const int i = blockIdx.x;
    const int f = threadIdx.x;
    hrow[f] = h[(size_t)i * IN_F + f];
    __syncthreads();
    float s = 0.f;
#pragma unroll 8
    for (int k = 0; k < IN_F; ++k)
        s = fmaf(hrow[k], W[k * OUT_F + f], s);
    Wh[(size_t)i * OUT_F + f] = s;
}

// ---------------------------------------------------------------------------
// Kernel 2: per-row scalars.
//   c[i] = Wh[i,:]·a2 + pos[i]·a3      (row term of e)
//   d[j] = Wh[j,:]·a1 - pos[j]·a3      (col term of e)
// ---------------------------------------------------------------------------
extern "C" __global__ void row_scalars(const float* __restrict__ Wh,
                                       const float* __restrict__ pos,
                                       const float* __restrict__ a,
                                       float* __restrict__ c,
                                       float* __restrict__ d, int N) {
    const int i = blockIdx.x * blockDim.x + threadIdx.x;
    if (i >= N) return;
    float sj = 0.f, si = 0.f;
#pragma unroll 8
    for (int f = 0; f < OUT_F; ++f) {
        float v = Wh[(size_t)i * OUT_F + f];
        sj = fmaf(v, a[f], sj);           // a1
        si = fmaf(v, a[OUT_F + f], si);   // a2
    }
    const float pa = pos[i * 3 + 0] * a[2 * OUT_F + 0] +
                     pos[i * 3 + 1] * a[2 * OUT_F + 1] +
                     pos[i * 3 + 2] * a[2 * OUT_F + 2];
    c[i] = si + pa;
    d[i] = sj - pa;
}

// ---------------------------------------------------------------------------
// Kernel 3: dmax = max_j d[j].  Monotonic leaky_relu => softmax row max is
// lrelu(c_i + dmax): one global scalar, so disjoint j-range partials combine
// by plain addition (enables the JSPLIT decomposition).
// ---------------------------------------------------------------------------
extern "C" __global__ void dmax_reduce(const float* __restrict__ d,
                                       float* __restrict__ dmax, int N) {
    __shared__ float red[256];
    float m = -3.402823466e38f;
    for (int j = threadIdx.x; j < N; j += 256) m = fmaxf(m, d[j]);
    red[threadIdx.x] = m;
    __syncthreads();
    for (int s = 128; s > 0; s >>= 1) {
        if (threadIdx.x < s)
            red[threadIdx.x] = fmaxf(red[threadIdx.x], red[threadIdx.x + s]);
        __syncthreads();
    }
    if (threadIdx.x == 0) dmax[0] = red[0];
}

// ---------------------------------------------------------------------------
// Kernel 4: pack Wh (f32, row-major) into f16 WMMA B-fragment order.
// B is the 32x16 (K x N) WMMA tile in this code's contiguous k order (make_a uses the same):
//   lanes 0-15  : N = lane,    K = 0..15  (pair q -> K = q)
//   lanes 16-31 : N = lane-16, K = 16..31 (pair q -> K = 16 + q)
// so the hot loop's B load is one contiguous 32B v16h per lane.
// ---------------------------------------------------------------------------
extern "C" __global__ void pack_whb(const float* __restrict__ Wh,
                                    _Float16* __restrict__ WhB, int total) {
    const int e = blockIdx.x * 256 + threadIdx.x;
    if (e >= total) return;
    const int q    = e & 15;
    const int lane = (e >> 4) & 31;
    const int ft   = (e >> 9) & 7;
    const int jc   = e >> 12;
    const int k = (lane & 16) + q;
    const int n = lane & 15;
    WhB[e] = (_Float16)Wh[(size_t)(jc * 32 + k) * OUT_F + ft * 16 + n];
}

// ---------------------------------------------------------------------------
// Pipelined helpers for the attention hot loop.
// ---------------------------------------------------------------------------
__device__ __forceinline__ void load_b(const _Float16* __restrict__ WhB,
                                       int jc, int lane, v16h B[8]) {
    const v16h* Bp = (const v16h*)WhB + (size_t)(jc * 8) * 32 + lane;
#pragma unroll
    for (int t = 0; t < 8; ++t) B[t] = Bp[t * 32];
}

__device__ __forceinline__ void load_d(const float* __restrict__ d,
                                       int j0, int koff, float2 dv[8]) {
#pragma unroll
    for (int p = 0; p < 8; ++p) {
        const int k = koff + 2 * p;
        dv[p] = *(const float2*)(d + j0 + k);   // k even -> 8B aligned
    }
}

__device__ __forceinline__ v16h make_a(const float2 dv[8], float cM, float mM,
                                       float& zacc) {
    v16h A;
#pragma unroll
    for (int p = 0; p < 8; ++p) {
        const float w0 = __builtin_amdgcn_exp2f((lrelu(cM + dv[p].x) - mM) * LOG2E);
        const float w1 = __builtin_amdgcn_exp2f((lrelu(cM + dv[p].y) - mM) * LOG2E);
        zacc += w0 + w1;
        A[2 * p]     = (_Float16)w0;
        A[2 * p + 1] = (_Float16)w1;
    }
    return A;
}

__device__ __forceinline__ void wmma8(v8f acc[8], v16h A, const v16h B[8]) {
#pragma unroll
    for (int t = 0; t < 8; ++t)
        acc[t] = __builtin_amdgcn_wmma_f32_16x16x32_f16(
            false, A, false, B[t], (short)0, acc[t], false, false);
}

// ---------------------------------------------------------------------------
// Kernel 5: fused attention partial with explicit software pipelining.
// Per iteration: issue loads for chunk jc+1, run 8 WMMAs on chunk jc (whose
// operands were loaded one full iteration earlier), then build A for jc+1
// (its load-wait sits after the WMMAs). Unrolled x2 with swapped buffer roles
// so no register copies are needed.
// ---------------------------------------------------------------------------
extern "C" __global__ __launch_bounds__(32) void attn_wmma_partial(
    const _Float16* __restrict__ WhB,
    const float* __restrict__ c,
    const float* __restrict__ d,
    const float* __restrict__ dmaxp,
    float* __restrict__ outraw,
    float* __restrict__ Zbuf, int N) {
    const int lane = threadIdx.x;
    const int i0   = blockIdx.x * 16;
    const int M    = lane & 15;
    const int hi   = lane >> 4;          // 0: lanes 0-15, 1: lanes 16-31
    const int koff = hi ? 16 : 0;

    const float cM = c[i0 + M];
    const float mM = lrelu(cM + dmaxp[0]);   // exact softmax row max

    v8f acc[8] = {};
    float zacc = 0.f;

    const int nchunks = N >> 5;
    const int cper    = nchunks / JSPLIT;    // 64 for N=8192 (even)
    const int jc0     = blockIdx.y * cper;
    const int jcend   = jc0 + cper;

    v16h   B0[8], B1[8];
    v16h   A0, A1;
    float2 dv[8];

    // ---- prologue: chunk jc0 ----
    load_b(WhB, jc0, lane, B0);
    load_d(d, jc0 << 5, koff, dv);
    A0 = make_a(dv, cM, mM, zacc);

    // ---- steady state: two chunks per iteration, buffers alternate ----
    for (int jc = jc0; jc + 2 < jcend; jc += 2) {
        load_b(WhB, jc + 1, lane, B1);
        load_d(d, (jc + 1) << 5, koff, dv);
        wmma8(acc, A0, B0);                  // operands loaded last iteration
        A1 = make_a(dv, cM, mM, zacc);

        load_b(WhB, jc + 2, lane, B0);
        load_d(d, (jc + 2) << 5, koff, dv);
        wmma8(acc, A1, B1);
        A0 = make_a(dv, cM, mM, zacc);
    }

    // ---- epilogue: chunks jcend-2 (in A0/B0) and jcend-1 ----
    load_b(WhB, jcend - 1, lane, B1);
    load_d(d, (jcend - 1) << 5, koff, dv);
    wmma8(acc, A0, B0);
    A1 = make_a(dv, cM, mM, zacc);
    wmma8(acc, A1, B1);

    // lane l and l+16 cover complementary K halves of row (l&15)
    const float ztot = zacc + __shfl_xor(zacc, 16);
    if (lane < 16) unsafeAtomicAdd(&Zbuf[i0 + lane], ztot);

    // C/D layout: lane holds col N=lane&15; VGPR r holds M=r (lanes 0-15)
    // or M=r+8 (lanes 16-31).
#pragma unroll
    for (int r = 0; r < 8; ++r) {
        const int row = r + (hi ? 8 : 0);
#pragma unroll
        for (int t = 0; t < 8; ++t) {
            unsafeAtomicAdd(&outraw[(size_t)(i0 + row) * OUT_F + t * 16 + M],
                            acc[t][r]);
        }
    }
}

// ---------------------------------------------------------------------------
// Kernel 6: normalize by Z and apply elu.
// ---------------------------------------------------------------------------
extern "C" __global__ void finalize(float* __restrict__ out,
                                    const float* __restrict__ Zbuf,
                                    int total) {
    const int t = blockIdx.x * 256 + threadIdx.x;
    if (t >= total) return;
    const float v = out[t] / Zbuf[t >> 7];           // row = t / 128
    out[t] = (v > 0.f) ? v : (__builtin_amdgcn_exp2f(v * LOG2E) - 1.f);
}

// ---------------------------------------------------------------------------
// Launch
// ---------------------------------------------------------------------------
extern "C" void kernel_launch(void* const* d_in, const int* in_sizes, int n_in,
                              void* d_out, int out_size, void* d_ws, size_t ws_size,
                              hipStream_t stream) {
    const float* h   = (const float*)d_in[0];   // (N, 128)
    const float* pos = (const float*)d_in[1];   // (N, 1, 3)
    const float* W   = (const float*)d_in[2];   // (128, 128)
    const float* a   = (const float*)d_in[3];   // (259, 1)
    float* out = (float*)d_out;                 // (N, 128) f32

    const int N = in_sizes[0] / IN_F;           // 8192

    char* ws = (char*)d_ws;
    float*    Wh   = (float*)ws;                                     // N*128*4 B
    _Float16* WhB  = (_Float16*)(ws + (size_t)N * OUT_F * 4);        // N*128*2 B
    float*    cbuf = (float*)(ws + (size_t)N * OUT_F * 6);           // N*4 B
    float*    dbuf = cbuf + N;                                       // N*4 B
    float*    dmax = dbuf + N;                                       // 4 B
    float*    Zbuf = dmax + 1;                                       // N*4 B

    const int total = N * OUT_F;

    zero_bufs<<<(total + 255) / 256, 256, 0, stream>>>(out, Zbuf, total, N);
    wh_gemm<<<N, IN_F, 0, stream>>>(h, W, Wh);
    row_scalars<<<(N + 255) / 256, 256, 0, stream>>>(Wh, pos, a, cbuf, dbuf, N);
    dmax_reduce<<<1, 256, 0, stream>>>(dbuf, dmax, N);
    pack_whb<<<(total + 255) / 256, 256, 0, stream>>>(Wh, WhB, total);
    attn_wmma_partial<<<dim3(N / 16, JSPLIT), 32, 0, stream>>>(
        WhB, cbuf, dbuf, dmax, out, Zbuf, N);
    finalize<<<(total + 255) / 256, 256, 0, stream>>>(out, Zbuf, total);
}
